// GATTopK2_72095321030887
// MI455X (gfx1250) — compile-verified, matrix operand order checked by reading, not run on hardware
//
#include <hip/hip_runtime.h>
#include <math.h>

// ---------------- problem constants (from reference) ----------------
#define NODES1 32768
#define EDGES  262144
#define INF_   128
#define FDIM   256
#define HEADS_ 4
#define BGR    32
#define NPG1   1024
#define KEEP1  512
#define KEEP2  256
#define OUTD   256
#define NODES2 16384   // BGR*KEEP1
#define NEGSLOPE 0.2f

typedef __attribute__((ext_vector_type(16))) _Float16 v16h;
typedef __attribute__((ext_vector_type(8)))  _Float16 v8h;
typedef __attribute__((ext_vector_type(8)))  float    v8f;

// order-preserving float<->uint encoding for atomic segment-max
__device__ __forceinline__ unsigned fenc(float f) {
    unsigned u = __float_as_uint(f);
    return (u & 0x80000000u) ? ~u : (u | 0x80000000u);
}
__device__ __forceinline__ float fdec(unsigned u) {
    return __uint_as_float((u & 0x80000000u) ? (u & 0x7fffffffu) : ~u);
}
__device__ __forceinline__ float gelu_exact(float x) {
    return 0.5f * x * (1.0f + erff(x * 0.70710678118654752440f));
}

// ---------------- f32 -> f16 conversion ----------------
__global__ void k_cvt_f16(const float* __restrict__ s, _Float16* __restrict__ d, int n) {
    int i = blockIdx.x * blockDim.x + threadIdx.x;
    if (i < n) d[i] = (_Float16)s[i];
}

// ---------------- WMMA GEMM: C[M,Nc] = A[M,K] @ B[K,Nc], f16 in, f32 out
// block = 256 threads = 8 waves; each block computes a 128x16 C tile.
// B column panel (K x 16) staged transposed in LDS once, shared by all waves.
// grid = (Nc/16, M/128)
template<int K>
__global__ __launch_bounds__(256)
void k_gemm_wmma(const _Float16* __restrict__ A, const _Float16* __restrict__ B,
                 float* __restrict__ C, int Nc) {
    __shared__ _Float16 bsh[K * 16];   // transposed: bsh[c*K + k]
    const int t  = threadIdx.x;
    const int n0 = blockIdx.x * 16;
    for (int i = t; i < K * 16; i += 256) {
        int k = i >> 4, c = i & 15;
        bsh[c * K + k] = B[(size_t)k * Nc + n0 + c];
    }
    __syncthreads();

    const int lane = t & 31, wave = t >> 5;
    const int hsel = lane >> 4;          // 0 or 1
    const int lrow = lane & 15;
    const int m0   = blockIdx.y * 128 + wave * 16;

    const _Float16* Arow = A + (size_t)(m0 + lrow) * K;
    const _Float16* Bcol = bsh + lrow * K;
    __builtin_prefetch(Arow + K, 0, 1);   // global_prefetch_b8: next row of A stream

    v8f acc = {};
#pragma unroll
    for (int k0 = 0; k0 < K; k0 += 32) {
        // lane fragment rows: K = k0 + hsel*8 + {0..7} and +16
        v8h alo = *(const v8h*)(Arow + k0 + hsel * 8);
        v8h ahi = *(const v8h*)(Arow + k0 + hsel * 8 + 16);
        v8h blo = *(const v8h*)(Bcol + k0 + hsel * 8);
        v8h bhi = *(const v8h*)(Bcol + k0 + hsel * 8 + 16);
        v16h a = __builtin_shufflevector(alo, ahi, 0,1,2,3,4,5,6,7,8,9,10,11,12,13,14,15);
        v16h b = __builtin_shufflevector(blo, bhi, 0,1,2,3,4,5,6,7,8,9,10,11,12,13,14,15);
        acc = __builtin_amdgcn_wmma_f32_16x16x32_f16(false, a, false, b,
                                                     (short)0, acc, false, false);
    }
    // D layout: VGPR r -> row = r + 8*hsel, col = lane&15
    const int col = n0 + lrow;
#pragma unroll
    for (int r = 0; r < 8; ++r)
        C[(size_t)(m0 + r + 8 * hsel) * Nc + col] = acc[r];
}

// ---------------- fills ----------------
__global__ void fill_f32(float* p, float v, int n) {
    int i = blockIdx.x * blockDim.x + threadIdx.x;
    if (i < n) p[i] = v;
}
__global__ void fill_u32(unsigned* p, unsigned v, int n) {
    int i = blockIdx.x * blockDim.x + threadIdx.x;
    if (i < n) p[i] = v;
}
__global__ void fill_i32(int* p, int v, int n) {
    int i = blockIdx.x * blockDim.x + threadIdx.x;
    if (i < n) p[i] = v;
}

// ---------------- attention coefficients: asn/adn[n,h] = sum_c h[n,h*64+c]*a[h,c]
__global__ void k_coef(const float* __restrict__ hbuf, const float* __restrict__ a_s,
                       const float* __restrict__ a_d, float* __restrict__ asn,
                       float* __restrict__ adn, int n) {
    int t = blockIdx.x * blockDim.x + threadIdx.x;
    if (t >= n * HEADS_) return;
    int nd = t >> 2, h = t & 3;
    const float* hr = hbuf + (size_t)nd * FDIM + h * 64;
    const float* as = a_s + h * 64;
    const float* ad = a_d + h * 64;
    float sa = 0.f, sd = 0.f;
    for (int c = 0; c < 64; ++c) { float v = hr[c]; sa += v * as[c]; sd += v * ad[c]; }
    asn[t] = sa; adn[t] = sd;
}

// ---------------- edge logits + segment max (atomic, encoded) ----------------
__global__ void k_logit_max(const int* __restrict__ src, const int* __restrict__ dst,
                            const int* __restrict__ vm, const float* __restrict__ asn,
                            const float* __restrict__ adn, float* __restrict__ lg,
                            unsigned* __restrict__ mxe, int Ereal, int ntot) {
    int e = blockIdx.x * blockDim.x + threadIdx.x;
    if (e >= Ereal + ntot) return;
    int s, d; bool valid;
    if (e < Ereal) { s = src[e]; d = dst[e]; valid = vm ? (vm[e] != 0) : true; }
    else           { s = d = e - Ereal; valid = true; }
#pragma unroll
    for (int h = 0; h < HEADS_; ++h) {
        float x = asn[s * HEADS_ + h] + adn[d * HEADS_ + h];
        float l = x > 0.f ? x : NEGSLOPE * x;
        lg[(size_t)e * HEADS_ + h] = l;
        if (valid) atomicMax(&mxe[d * HEADS_ + h], fenc(l));
    }
}

// ---------------- p = exp(logit - max), segment-sum denominator ----------------
__global__ void k_exp_den(const int* __restrict__ src, const int* __restrict__ dst,
                          const int* __restrict__ vm, float* __restrict__ lg,
                          const unsigned* __restrict__ mxe, float* __restrict__ den,
                          int Ereal, int ntot) {
    int e = blockIdx.x * blockDim.x + threadIdx.x;
    if (e >= Ereal + ntot) return;
    int d; bool valid;
    if (e < Ereal) { d = dst[e]; valid = vm ? (vm[e] != 0) : true; }
    else           { d = e - Ereal; valid = true; }
#pragma unroll
    for (int h = 0; h < HEADS_; ++h) {
        size_t idx = (size_t)e * HEADS_ + h;
        float p = 0.f;
        if (valid) {
            p = expf(lg[idx] - fdec(mxe[d * HEADS_ + h]));
            atomicAdd(&den[d * HEADS_ + h], p);
        }
        lg[idx] = p;  // overwrite logit with p
    }
}

// ---------------- weighted aggregation: out[d,:] += alpha * h[s,:] ----------------
__global__ void k_aggregate(const int* __restrict__ src, const int* __restrict__ dst,
                            const int* __restrict__ vm, const float* __restrict__ p,
                            const float* __restrict__ den, const float* __restrict__ hbuf,
                            float* __restrict__ outp, int Ereal, int ntot) {
    int t = blockIdx.x * blockDim.x + threadIdx.x;
    int e = t >> 6, j = t & 63;  // 64 lanes per edge
    if (e >= Ereal + ntot) return;
    int s, d; bool valid;
    if (e < Ereal) { s = src[e]; d = dst[e]; valid = vm ? (vm[e] != 0) : true; }
    else           { s = d = e - Ereal; valid = true; }
    if (!valid) return;
#pragma unroll
    for (int h = 0; h < HEADS_; ++h) {
        float alpha = p[(size_t)e * HEADS_ + h] / den[d * HEADS_ + h];
        int ch = h * 64 + j;
        atomicAdd(&outp[(size_t)d * FDIM + ch], alpha * hbuf[(size_t)s * FDIM + ch]);
    }
}

// ---------------- bias + exact GELU ----------------
__global__ void k_bias_gelu(float* y, const float* __restrict__ bias, int total) {
    int i = blockIdx.x * blockDim.x + threadIdx.x;
    if (i >= total) return;
    float x = y[i] + bias[i & (FDIM - 1)];
    y[i] = gelu_exact(x);
}

// ---------------- BN stats: one block per channel ----------------
__global__ void k_bn_stats(const float* __restrict__ y, float* mu, float* rsig, int rows) {
    int c = blockIdx.x, t = threadIdx.x;
    float s = 0.f, q = 0.f;
    for (int r = t; r < rows; r += blockDim.x) {
        float v = y[(size_t)r * FDIM + c];
        s += v; q += v * v;
    }
    __shared__ float ss[256], sq[256];
    ss[t] = s; sq[t] = q; __syncthreads();
    for (int w = 128; w > 0; w >>= 1) {
        if (t < w) { ss[t] += ss[t + w]; sq[t] += sq[t + w]; }
        __syncthreads();
    }
    if (t == 0) {
        float m = ss[0] / rows;
        float v = sq[0] / rows - m * m;
        mu[c] = m; rsig[c] = rsqrtf(v + 1e-5f);
    }
}
__global__ void k_bn_apply(float* y, const float* __restrict__ mu, const float* __restrict__ rsig,
                           const float* __restrict__ g, const float* __restrict__ be, int total) {
    int i = blockIdx.x * blockDim.x + threadIdx.x;
    if (i >= total) return;
    int c = i & (FDIM - 1);
    y[i] = (y[i] - mu[c]) * rsig[c] * g[c] + be[c];
}

// ---------------- ||w|| ----------------
__global__ void k_wnorm(const float* __restrict__ w, float* outv) {
    __shared__ float ss[256];
    int t = threadIdx.x;
    float v = w[t];
    ss[t] = v * v; __syncthreads();
    for (int k = 128; k > 0; k >>= 1) {
        if (t < k) ss[t] += ss[t + k];
        __syncthreads();
    }
    if (t == 0) outv[0] = sqrtf(ss[0]);
}

// ---------------- pooling score ----------------
__global__ void k_score(const float* __restrict__ x, const float* __restrict__ w,
                        const float* __restrict__ nrm, float* score, int n) {
    int i = blockIdx.x * blockDim.x + threadIdx.x;
    if (i >= n) return;
    const float* r = x + (size_t)i * FDIM;
    float s = 0.f;
    for (int c = 0; c < FDIM; ++c) s += r[c] * w[c];
    score[i] = s / nrm[0];
}

// ---------------- per-graph top-k via in-LDS bitonic sort (desc, idx tiebreak) ----
__global__ void k_topk(const float* __restrict__ score, int npg, int kkeep,
                       int* gidx, float* tanhv, int* inv) {
    int g = blockIdx.x, t = threadIdx.x;
    __shared__ float sv[1024];
    __shared__ int   si[1024];
    for (int i = t; i < npg; i += blockDim.x) { sv[i] = score[g * npg + i]; si[i] = i; }
    __syncthreads();
    for (int k = 2; k <= npg; k <<= 1) {
        for (int j = k >> 1; j > 0; j >>= 1) {
            for (int i = t; i < npg; i += blockDim.x) {
                int l = i ^ j;
                if (l > i) {
                    bool dir = ((i & k) == 0);  // true -> descending block
                    float vi = sv[i], vl = sv[l];
                    int ii = si[i], il = si[l];
                    bool wrong = dir ? (vi < vl || (vi == vl && ii > il))
                                     : (vi > vl || (vi == vl && ii < il));
                    if (wrong) { sv[i] = vl; sv[l] = vi; si[i] = il; si[l] = ii; }
                }
            }
            __syncthreads();
        }
    }
    for (int i = t; i < kkeep; i += blockDim.x) {
        int idxg = g * npg + si[i];
        int pos  = g * kkeep + i;
        gidx[pos]  = idxg;
        tanhv[pos] = tanhf(sv[i]);
        inv[idxg]  = pos;
    }
}

// ---------------- gather kept nodes, scaled by tanh(score) ----------------
__global__ void k_gather(const float* __restrict__ x, const int* __restrict__ gidx,
                         const float* __restrict__ tanhv, float* xn, int total) {
    int idx = blockIdx.x * blockDim.x + threadIdx.x;
    if (idx >= total) return;
    int i = idx >> 8, ch = idx & (FDIM - 1);
    xn[idx] = x[(size_t)gidx[i] * FDIM + ch] * tanhv[i];
}

// ---------------- filter_adj: remap edges through inv ----------------
__global__ void k_edgefilt(const int* __restrict__ src, const int* __restrict__ dst,
                           const int* __restrict__ inv, int* s2, int* d2, int* vm, int E) {
    int e = blockIdx.x * blockDim.x + threadIdx.x;
    if (e >= E) return;
    int sn = inv[src[e]], dn = inv[dst[e]];
    int valid = (sn >= 0 && dn >= 0) ? 1 : 0;
    vm[e] = valid;
    s2[e] = valid ? sn : 0;
    d2[e] = valid ? dn : 0;
}

// ---------------- readout: [gmp | gap]; optional gather+scale source ----------------
__global__ void k_readout(const float* __restrict__ x, const int* __restrict__ gidx,
                          const float* __restrict__ tanhv, int kk, float* outr) {
    int t = blockIdx.x * blockDim.x + threadIdx.x;
    if (t >= BGR * FDIM) return;
    int b = t >> 8, ch = t & (FDIM - 1);
    float mx = -3.4028235e38f, sm = 0.f;
    for (int i = 0; i < kk; ++i) {
        int row = b * kk + i;
        float v;
        if (gidx) v = x[(size_t)gidx[row] * FDIM + ch] * tanhv[row];
        else      v = x[(size_t)row * FDIM + ch];
        mx = fmaxf(mx, v); sm += v;
    }
    outr[(size_t)b * (2 * FDIM) + ch]        = mx;
    outr[(size_t)b * (2 * FDIM) + FDIM + ch] = sm / kk;
}

// ---------------- final linear: out = (x1+x2) @ Wl^T + bl ----------------
__global__ void k_final(const float* __restrict__ x1, const float* __restrict__ x2,
                        const float* __restrict__ Wl, const float* __restrict__ bl,
                        float* out) {
    int t = blockIdx.x * blockDim.x + threadIdx.x;
    if (t >= BGR * OUTD) return;
    int b = t >> 8, o = t & (OUTD - 1);
    float acc = bl[o];
    const float* wr = Wl + (size_t)o * (2 * FDIM);
    const float* r1 = x1 + (size_t)b * (2 * FDIM);
    const float* r2 = x2 + (size_t)b * (2 * FDIM);
    for (int c = 0; c < 2 * FDIM; ++c) acc += (r1[c] + r2[c]) * wr[c];
    out[t] = acc;
}

// =====================================================================
extern "C" void kernel_launch(void* const* d_in, const int* in_sizes, int n_in,
                              void* d_out, int out_size, void* d_ws, size_t ws_size,
                              hipStream_t stream) {
    (void)in_sizes; (void)n_in; (void)out_size; (void)ws_size;
    const float* x   = (const float*)d_in[0];
    const int*   ei  = (const int*)d_in[1];
    const float* W1  = (const float*)d_in[3];
    const float* as1 = (const float*)d_in[4];
    const float* ad1 = (const float*)d_in[5];
    const float* b1  = (const float*)d_in[6];
    const float* g1  = (const float*)d_in[7];
    const float* be1 = (const float*)d_in[8];
    const float* pw1 = (const float*)d_in[9];
    const float* W2  = (const float*)d_in[10];
    const float* as2 = (const float*)d_in[11];
    const float* ad2 = (const float*)d_in[12];
    const float* b2  = (const float*)d_in[13];
    const float* g2  = (const float*)d_in[14];
    const float* be2 = (const float*)d_in[15];
    const float* pw2 = (const float*)d_in[16];
    const float* Wl  = (const float*)d_in[17];
    const float* bl  = (const float*)d_in[18];
    const int* src = ei;
    const int* dst = ei + EDGES;

    // bump-allocated workspace (buffers reused across layers)
    char* base = (char*)d_ws;
    size_t off = 0;
    auto alloc = [&](size_t bytes) -> char* {
        off = (off + 255) & ~(size_t)255;
        char* p = base + off; off += bytes; return p;
    };
    float*    hbuf  = (float*)   alloc((size_t)NODES1 * FDIM * 4);     // h1 / h2
    float*    obuf  = (float*)   alloc((size_t)NODES1 * FDIM * 4);     // out1 / out2
    float*    asn   = (float*)   alloc((size_t)NODES1 * HEADS_ * 4);
    float*    adn   = (float*)   alloc((size_t)NODES1 * HEADS_ * 4);
    unsigned* mxe   = (unsigned*)alloc((size_t)NODES1 * HEADS_ * 4);
    float*    den   = (float*)   alloc((size_t)NODES1 * HEADS_ * 4);
    float*    lg    = (float*)   alloc((size_t)(EDGES + NODES1) * HEADS_ * 4);
    float*    xn1   = (float*)   alloc((size_t)NODES2 * FDIM * 4);
    _Float16* xh    = (_Float16*)alloc((size_t)NODES1 * INF_ * 2);     // f16 A (layer1 x / layer2 xn1)
    _Float16* wh    = (_Float16*)alloc((size_t)FDIM * FDIM * 2);       // f16 W
    float*    score = (float*)   alloc((size_t)NODES1 * 4);
    int*      inv   = (int*)     alloc((size_t)NODES1 * 4);
    int*      gidx1 = (int*)     alloc((size_t)BGR * KEEP1 * 4);
    float*    tanh1 = (float*)   alloc((size_t)BGR * KEEP1 * 4);
    int*      gidx2 = (int*)     alloc((size_t)BGR * KEEP2 * 4);
    float*    tanh2 = (float*)   alloc((size_t)BGR * KEEP2 * 4);
    int*      s2    = (int*)     alloc((size_t)EDGES * 4);
    int*      d2    = (int*)     alloc((size_t)EDGES * 4);
    int*      vm2   = (int*)     alloc((size_t)EDGES * 4);
    float*    x1r   = (float*)   alloc((size_t)BGR * 2 * FDIM * 4);
    float*    x2r   = (float*)   alloc((size_t)BGR * 2 * FDIM * 4);
    float*    mu    = (float*)   alloc(FDIM * 4);
    float*    rsig  = (float*)   alloc(FDIM * 4);
    float*    wn    = (float*)   alloc(2 * 4);

    auto nb = [](long n, int b) { return dim3((unsigned)((n + b - 1) / b)); };

    // ================= layer 1 =================
    k_wnorm<<<1, 256, 0, stream>>>(pw1, wn + 0);
    k_cvt_f16<<<nb((long)NODES1 * INF_, 256), 256, 0, stream>>>(x, xh, NODES1 * INF_);
    k_cvt_f16<<<nb(INF_ * FDIM, 256), 256, 0, stream>>>(W1, wh, INF_ * FDIM);
    k_gemm_wmma<INF_><<<dim3(FDIM / 16, NODES1 / 128), 256, 0, stream>>>(xh, wh, hbuf, FDIM);
    k_coef<<<nb((long)NODES1 * HEADS_, 256), 256, 0, stream>>>(hbuf, as1, ad1, asn, adn, NODES1);
    fill_u32<<<nb(NODES1 * HEADS_, 256), 256, 0, stream>>>(mxe, 0u, NODES1 * HEADS_);
    fill_f32<<<nb(NODES1 * HEADS_, 256), 256, 0, stream>>>(den, 0.f, NODES1 * HEADS_);
    k_logit_max<<<nb(EDGES + NODES1, 256), 256, 0, stream>>>(src, dst, nullptr, asn, adn, lg, mxe, EDGES, NODES1);
    k_exp_den<<<nb(EDGES + NODES1, 256), 256, 0, stream>>>(src, dst, nullptr, lg, mxe, den, EDGES, NODES1);
    fill_f32<<<nb((long)NODES1 * FDIM, 256), 256, 0, stream>>>(obuf, 0.f, NODES1 * FDIM);
    k_aggregate<<<nb((long)(EDGES + NODES1) * 64, 256), 256, 0, stream>>>(src, dst, nullptr, lg, den, hbuf, obuf, EDGES, NODES1);
    k_bias_gelu<<<nb((long)NODES1 * FDIM, 256), 256, 0, stream>>>(obuf, b1, NODES1 * FDIM);
    k_bn_stats<<<FDIM, 256, 0, stream>>>(obuf, mu, rsig, NODES1);
    k_bn_apply<<<nb((long)NODES1 * FDIM, 256), 256, 0, stream>>>(obuf, mu, rsig, g1, be1, NODES1 * FDIM);
    k_score<<<nb(NODES1, 256), 256, 0, stream>>>(obuf, pw1, wn + 0, score, NODES1);
    fill_i32<<<nb(NODES1, 256), 256, 0, stream>>>(inv, -1, NODES1);
    k_topk<<<BGR, 256, 0, stream>>>(score, NPG1, KEEP1, gidx1, tanh1, inv);
    k_gather<<<nb((long)NODES2 * FDIM, 256), 256, 0, stream>>>(obuf, gidx1, tanh1, xn1, NODES2 * FDIM);
    k_edgefilt<<<nb(EDGES, 256), 256, 0, stream>>>(src, dst, inv, s2, d2, vm2, EDGES);
    k_readout<<<nb(BGR * FDIM, 256), 256, 0, stream>>>(xn1, nullptr, nullptr, KEEP1, x1r);

    // ================= layer 2 =================
    k_wnorm<<<1, 256, 0, stream>>>(pw2, wn + 1);
    k_cvt_f16<<<nb((long)NODES2 * FDIM, 256), 256, 0, stream>>>(xn1, xh, NODES2 * FDIM);
    k_cvt_f16<<<nb(FDIM * FDIM, 256), 256, 0, stream>>>(W2, wh, FDIM * FDIM);
    k_gemm_wmma<FDIM><<<dim3(FDIM / 16, NODES2 / 128), 256, 0, stream>>>(xh, wh, hbuf, FDIM);
    k_coef<<<nb((long)NODES2 * HEADS_, 256), 256, 0, stream>>>(hbuf, as2, ad2, asn, adn, NODES2);
    fill_u32<<<nb(NODES2 * HEADS_, 256), 256, 0, stream>>>(mxe, 0u, NODES2 * HEADS_);
    fill_f32<<<nb(NODES2 * HEADS_, 256), 256, 0, stream>>>(den, 0.f, NODES2 * HEADS_);
    k_logit_max<<<nb(EDGES + NODES2, 256), 256, 0, stream>>>(s2, d2, vm2, asn, adn, lg, mxe, EDGES, NODES2);
    k_exp_den<<<nb(EDGES + NODES2, 256), 256, 0, stream>>>(s2, d2, vm2, lg, mxe, den, EDGES, NODES2);
    fill_f32<<<nb((long)NODES2 * FDIM, 256), 256, 0, stream>>>(obuf, 0.f, NODES2 * FDIM);
    k_aggregate<<<nb((long)(EDGES + NODES2) * 64, 256), 256, 0, stream>>>(s2, d2, vm2, lg, den, hbuf, obuf, EDGES, NODES2);
    k_bias_gelu<<<nb((long)NODES2 * FDIM, 256), 256, 0, stream>>>(obuf, b2, NODES2 * FDIM);
    k_bn_stats<<<FDIM, 256, 0, stream>>>(obuf, mu, rsig, NODES2);
    k_bn_apply<<<nb((long)NODES2 * FDIM, 256), 256, 0, stream>>>(obuf, mu, rsig, g2, be2, NODES2 * FDIM);
    k_score<<<nb(NODES2, 256), 256, 0, stream>>>(obuf, pw2, wn + 1, score, NODES2);
    fill_i32<<<nb(NODES2, 256), 256, 0, stream>>>(inv, -1, NODES2);
    k_topk<<<BGR, 256, 0, stream>>>(score, KEEP1 /*npg=512*/, KEEP2, gidx2, tanh2, inv);
    k_readout<<<nb(BGR * FDIM, 256), 256, 0, stream>>>(obuf, gidx2, tanh2, KEEP2, x2r);

    // ================= head =================
    k_final<<<nb(BGR * OUTD, 256), 256, 0, stream>>>(x1r, x2r, Wl, bl, (float*)d_out);
}
